// DeepseekV3MoE_19550691131495
// MI455X (gfx1250) — matrix operand order repaired and checked by reading, not compiled, not run on hardware
//
#include <hip/hip_runtime.h>
#include <hip/hip_bf16.h>

// ---------------------------------------------------------------------------
// DeepSeek-V3 MoE block for MI455X (gfx1250), wave32 + WMMA bf16.
// Pipeline: convert/transpose weights to bf16 [N][K] -> router (top-2 sigmoid)
// -> deterministic ballot/prefix compaction into packed expert lists
// -> dual-GEMM SwiGLU up (v_wmma_f32_16x16x32_bf16) -> down GEMM -> combine.
// Wave tile 32(M)x64(N); B fragments double-buffered one j-step ahead so each
// load's latency is covered by four in-flight WMMAs.
// ---------------------------------------------------------------------------

typedef __bf16 bf16_t;
typedef __attribute__((ext_vector_type(8)))  bf16_t v8bf;
typedef __attribute__((ext_vector_type(16))) bf16_t v16bf;
typedef __attribute__((ext_vector_type(8)))  float  v8f;

constexpr int T_TOK = 2048;   // tokens
constexpr int DIM   = 1024;   // model dim
constexpr int NEXP  = 8;      // experts
constexpr int HEXP  = 512;    // routed hidden
constexpr int HSH   = 1024;   // shared hidden

// ---------------- fragment helpers (layouts per CDNA5 ISA) -------

__device__ __forceinline__ v16bf concat16(v8bf lo, v8bf hi) {
  v16bf r;
#pragma unroll
  for (int i = 0; i < 8; ++i) { r[i] = lo[i]; r[i + 8] = hi[i]; }
  return r;
}

// A (16x32 bf16, row-major source, ld = Kdim): lane kg=lane>>4 holds
// K in [k+kg*8, k+kg*8+8) and [k+16+kg*8, ...+8) of row lane&15.
__device__ __forceinline__ v16bf load_a_frag(const bf16_t* __restrict__ row,
                                             int k, int kg) {
  v8bf lo = *(const v8bf*)(row + k + kg * 8);
  v8bf hi = *(const v8bf*)(row + k + 16 + kg * 8);
  return concat16(lo, hi);
}

// B (32x16 bf16) from transposed weights W[N][K]: lane holds column lane&15,
// K in [k+kg*8, k+kg*8+8) and [k+16+kg*8, ...+8) (same order as A).
__device__ __forceinline__ v16bf load_b_frag(const bf16_t* __restrict__ nrow,
                                             int k, int kg) {
  const bf16_t* p = nrow + k + kg * 8;
  v8bf lo = *(const v8bf*)(p);
  v8bf hi = *(const v8bf*)(p + 16);
  return concat16(lo, hi);
}

__device__ __forceinline__ v8f wmma_bf16(v16bf a, v16bf b, v8f c) {
  return __builtin_amdgcn_wmma_f32_16x16x32_bf16(false, a, false, b,
                                                 (short)0, c, false, false);
}

// ---------------- conversion kernels ---------------------------------------

__global__ void convert_f32_to_bf16(const float* __restrict__ in,
                                    bf16_t* __restrict__ out, int n) {
  int i = (blockIdx.x * blockDim.x + threadIdx.x) * 4;
  if (i + 3 < n) {
    float4 v = *(const float4*)(in + i);
    out[i + 0] = (bf16_t)v.x;
    out[i + 1] = (bf16_t)v.y;
    out[i + 2] = (bf16_t)v.z;
    out[i + 3] = (bf16_t)v.w;
  }
}

// in: [batch][R][C] fp32  ->  out: [batch][C][R] bf16   (LDS tiled)
__global__ void transpose_f32_to_bf16(const float* __restrict__ in,
                                      bf16_t* __restrict__ out, int R, int C) {
  __shared__ float tile[32][33];
  const float* ip = in + (size_t)blockIdx.z * R * C;
  bf16_t*      op = out + (size_t)blockIdx.z * R * C;
  int c0 = blockIdx.x * 32, r0 = blockIdx.y * 32;
#pragma unroll
  for (int i = threadIdx.y; i < 32; i += 8) {
    int r = r0 + i, c = c0 + threadIdx.x;
    if (r < R && c < C) tile[i][threadIdx.x] = ip[(size_t)r * C + c];
  }
  __syncthreads();
#pragma unroll
  for (int i = threadIdx.y; i < 32; i += 8) {
    int c = c0 + i, r = r0 + threadIdx.x;
    if (c < C && r < R) op[(size_t)c * R + r] = (bf16_t)tile[threadIdx.x][i];
  }
}

// ---------------- router: sigmoid scores, top-2, normalized ------------------

__global__ void router_kernel(const float* __restrict__ x,
                              const float* __restrict__ gate_w,
                              int* __restrict__ top_idx,
                              float* __restrict__ top_w) {
  int lane = threadIdx.x & 31;
  int t = blockIdx.x * (blockDim.x >> 5) + (threadIdx.x >> 5);
  if (t >= T_TOK) return;
  const float* xr = x + (size_t)t * DIM;
  float acc[NEXP];
#pragma unroll
  for (int e = 0; e < NEXP; ++e) acc[e] = 0.f;
  for (int d = lane; d < DIM; d += 32) {
    float xv = xr[d];
    const float* g = gate_w + (size_t)d * NEXP;
#pragma unroll
    for (int e = 0; e < NEXP; ++e) acc[e] += xv * g[e];
  }
#pragma unroll
  for (int e = 0; e < NEXP; ++e)
#pragma unroll
    for (int off = 16; off > 0; off >>= 1)
      acc[e] += __shfl_xor(acc[e], off, 32);
  if (lane == 0) {
    float s[NEXP];
#pragma unroll
    for (int e = 0; e < NEXP; ++e) s[e] = 1.f / (1.f + __expf(-acc[e]));
    int i1 = 0;
#pragma unroll
    for (int e = 1; e < NEXP; ++e) if (s[e] > s[i1]) i1 = e;
    int i2 = -1;
#pragma unroll
    for (int e = 0; e < NEXP; ++e)
      if (e != i1 && (i2 < 0 || s[e] > s[i2])) i2 = e;
    float sum = s[i1] + s[i2];
    top_idx[t * 2 + 0] = i1;
    top_idx[t * 2 + 1] = i2;
    top_w[t * 2 + 0] = s[i1] / sum;
    top_w[t * 2 + 1] = s[i2] / sum;
  }
}

// ---------------- deterministic compaction (one wave, ballot+prefix) --------

__global__ void assign_kernel(const int* __restrict__ top_idx,
                              const float* __restrict__ top_w,
                              int* __restrict__ tok_list,
                              float* __restrict__ wgt_list,
                              int* __restrict__ rowof,
                              int* __restrict__ counts,
                              int* __restrict__ offs) {
  int lane = threadIdx.x;  // blockDim.x == 32, single block
  int base = 0;
  for (int e = 0; e < NEXP; ++e) {
    int cnt = 0;
    for (int c = 0; c < T_TOK; c += 32) {
      int t = c + lane;
      int k = -1;
      int i0 = top_idx[t * 2 + 0], i1 = top_idx[t * 2 + 1];
      if (i0 == e) k = 0; else if (i1 == e) k = 1;
      unsigned long long m = __ballot(k >= 0);
      int pos = cnt + __popcll(m & ((1ull << lane) - 1ull));
      if (k >= 0) {
        int g = base + pos;
        tok_list[g] = t;
        wgt_list[g] = top_w[t * 2 + k];
        rowof[t * 2 + k] = g;
      }
      cnt += __popcll(m);
    }
    if (lane == 0) { counts[e] = cnt; offs[e] = base; }
    base += cnt;
  }
}

// ---------------- SwiGLU up-projection (dual GEMM, 32x64 wave tile) ---------
// C1 = X@W1, C3 = X@W3, H = silu(C1)*C3 (bf16 out).
// Block tile 64(M) x 256(N): 8 waves (2 along M, 4 along N); each wave holds
// two A fragments so every B fragment feeds two WMMAs; B double-buffered.

__global__ void __launch_bounds__(256)
moe_up_kernel(const bf16_t* __restrict__ Xb,      // [T][Kdim]
              const bf16_t* __restrict__ W1t,     // [e][Ndim][Kdim]
              const bf16_t* __restrict__ W3t,
              bf16_t* __restrict__ Hout,          // [rows][Ndim], packed
              const int* __restrict__ tok_list,   // null -> shared expert
              const int* __restrict__ counts,
              const int* __restrict__ offs,
              int Kdim, int Ndim) {
  const int e = blockIdx.z;
  const int nrows = counts ? counts[e] : T_TOK;
  const int gbase = offs ? offs[e] : 0;
  const int m_base = blockIdx.x * 64;
  if (m_base >= nrows) return;
  const int wave = threadIdx.x >> 5, lane = threadIdx.x & 31;
  const int row0 = m_base + (wave & 1) * 32;     // wave M strip: 32 rows
  if (row0 >= nrows) return;                     // wave-uniform exit
  const int col0 = blockIdx.y * 256 + (wave >> 1) * 64;
  const int kg = lane >> 4, ncol = lane & 15;
  int r0 = row0 + (lane & 15);      if (r0 > nrows - 1) r0 = nrows - 1;
  int r1 = row0 + 16 + (lane & 15); if (r1 > nrows - 1) r1 = nrows - 1;
  const int tok0 = tok_list ? tok_list[gbase + r0] : r0;
  const int tok1 = tok_list ? tok_list[gbase + r1] : r1;
  const bf16_t* Ap0 = Xb + (size_t)tok0 * Kdim;
  const bf16_t* Ap1 = Xb + (size_t)tok1 * Kdim;
  const bf16_t* B1r = W1t + (size_t)e * Ndim * Kdim + (size_t)(col0 + ncol) * Kdim;
  const bf16_t* B3r = W3t + (size_t)e * Ndim * Kdim + (size_t)(col0 + ncol) * Kdim;
  const size_t nstride = (size_t)16 * Kdim;      // next 16-column B row group

  v8f c1[2][4] = {};
  v8f c3[2][4] = {};
  for (int k = 0; k < Kdim; k += 32) {
    v16bf a0 = load_a_frag(Ap0, k, kg);
    v16bf a1 = load_a_frag(Ap1, k, kg);
    if (k + 32 < Kdim) {                        // global_prefetch_b8 (near)
      __builtin_prefetch(Ap0 + k + 32, 0, 3);
      __builtin_prefetch(Ap1 + k + 32, 0, 3);
    }
    // one-step rotation: preload j+1 B fragments before j's WMMAs
    v16bf b1 = load_b_frag(B1r, k, kg);
    v16bf b3 = load_b_frag(B3r, k, kg);
#pragma unroll
    for (int j = 0; j < 4; ++j) {
      v16bf nb1 = b1, nb3 = b3;
      if (j < 3) {
        nb1 = load_b_frag(B1r + (size_t)(j + 1) * nstride, k, kg);
        nb3 = load_b_frag(B3r + (size_t)(j + 1) * nstride, k, kg);
      }
      c1[0][j] = wmma_bf16(a0, b1, c1[0][j]);
      c1[1][j] = wmma_bf16(a1, b1, c1[1][j]);
      c3[0][j] = wmma_bf16(a0, b3, c3[0][j]);
      c3[1][j] = wmma_bf16(a1, b3, c3[1][j]);
      b1 = nb1; b3 = nb3;
    }
  }
#pragma unroll
  for (int mi = 0; mi < 2; ++mi)
#pragma unroll
    for (int j = 0; j < 4; ++j)
#pragma unroll
      for (int rr = 0; rr < 8; ++rr) {
        int grow = row0 + mi * 16 + kg * 8 + rr;   // C/D layout: M = kg*8 + r
        if (grow < nrows) {
          float u = c1[mi][j][rr], v = c3[mi][j][rr];
          float hv = (u / (1.f + __expf(-u))) * v;
          Hout[(size_t)(gbase + grow) * Ndim + (col0 + j * 16 + ncol)] =
              (bf16_t)hv;
        }
      }
}

// ---------------- down-projection (32x64 wave tile) --------------------------
// shared (wgt_list==null): Out[t][D] = H@W2   (plain stores, initializes out)
// routed: Yout[g][D] = (H@W2) * routing_weight[g]   (packed rows)

__global__ void __launch_bounds__(256)
moe_down_kernel(const bf16_t* __restrict__ Hin,   // [rows][Kdim] packed
                const bf16_t* __restrict__ W2t,   // [e][Ndim][Kdim]
                float* __restrict__ Yout,
                const int* __restrict__ counts,
                const int* __restrict__ offs,
                const float* __restrict__ wgt_list,
                int Kdim, int Ndim) {
  const int e = blockIdx.z;
  const int nrows = counts ? counts[e] : T_TOK;
  const int gbase = offs ? offs[e] : 0;
  const int m_base = blockIdx.x * 64;
  if (m_base >= nrows) return;
  const int wave = threadIdx.x >> 5, lane = threadIdx.x & 31;
  const int row0 = m_base + (wave & 1) * 32;
  if (row0 >= nrows) return;
  const int col0 = blockIdx.y * 256 + (wave >> 1) * 64;
  const int kg = lane >> 4, ncol = lane & 15;
  int r0 = row0 + (lane & 15);      if (r0 > nrows - 1) r0 = nrows - 1;
  int r1 = row0 + 16 + (lane & 15); if (r1 > nrows - 1) r1 = nrows - 1;
  const bf16_t* Ap0 = Hin + (size_t)(gbase + r0) * Kdim;
  const bf16_t* Ap1 = Hin + (size_t)(gbase + r1) * Kdim;
  const bf16_t* B2r = W2t + (size_t)e * Ndim * Kdim + (size_t)(col0 + ncol) * Kdim;
  const size_t nstride = (size_t)16 * Kdim;

  v8f c[2][4] = {};
  for (int k = 0; k < Kdim; k += 32) {
    v16bf a0 = load_a_frag(Ap0, k, kg);
    v16bf a1 = load_a_frag(Ap1, k, kg);
    if (k + 32 < Kdim) {
      __builtin_prefetch(Ap0 + k + 32, 0, 3);
      __builtin_prefetch(Ap1 + k + 32, 0, 3);
    }
    v16bf b = load_b_frag(B2r, k, kg);
#pragma unroll
    for (int j = 0; j < 4; ++j) {
      v16bf nb = b;
      if (j < 3) nb = load_b_frag(B2r + (size_t)(j + 1) * nstride, k, kg);
      c[0][j] = wmma_bf16(a0, b, c[0][j]);
      c[1][j] = wmma_bf16(a1, b, c[1][j]);
      b = nb;
    }
  }
#pragma unroll
  for (int mi = 0; mi < 2; ++mi)
#pragma unroll
    for (int j = 0; j < 4; ++j)
#pragma unroll
      for (int rr = 0; rr < 8; ++rr) {
        int grow = row0 + mi * 16 + kg * 8 + rr;
        if (grow < nrows) {
          size_t col = (size_t)(col0 + j * 16 + ncol);
          if (wgt_list) {
            float w = wgt_list[gbase + grow];
            Yout[(size_t)(gbase + grow) * Ndim + col] = c[mi][j][rr] * w;
          } else {
            Yout[(size_t)grow * Ndim + col] = c[mi][j][rr];
          }
        }
      }
}

// ---------------- final combine: out[t] += y[row0(t)] + y[row1(t)] ----------

__global__ void combine_kernel(const float* __restrict__ ybuf,
                               const int* __restrict__ rowof,
                               float* __restrict__ out) {
  int i = blockIdx.x * blockDim.x + threadIdx.x;       // over T*D/4
  int t = i / (DIM / 4);
  int d4 = i % (DIM / 4);
  int r0 = rowof[t * 2 + 0], r1 = rowof[t * 2 + 1];
  float4 a = *((const float4*)(out + (size_t)t * DIM) + d4);
  float4 b = *((const float4*)(ybuf + (size_t)r0 * DIM) + d4);
  float4 c = *((const float4*)(ybuf + (size_t)r1 * DIM) + d4);
  a.x += b.x + c.x; a.y += b.y + c.y; a.z += b.z + c.z; a.w += b.w + c.w;
  *((float4*)(out + (size_t)t * DIM) + d4) = a;
}

// ---------------------------------------------------------------------------

extern "C" void kernel_launch(void* const* d_in, const int* in_sizes, int n_in,
                              void* d_out, int out_size, void* d_ws, size_t ws_size,
                              hipStream_t stream) {
  (void)in_sizes; (void)n_in; (void)out_size; (void)ws_size;
  const float* x      = (const float*)d_in[0];
  const float* gate_w = (const float*)d_in[1];
  const float* w1     = (const float*)d_in[2];
  const float* w3     = (const float*)d_in[3];
  const float* w2     = (const float*)d_in[4];
  const float* ws1    = (const float*)d_in[5];
  const float* ws3    = (const float*)d_in[6];
  const float* ws2    = (const float*)d_in[7];
  float* out = (float*)d_out;

  char* ws = (char*)d_ws;
  size_t off = 0;
  auto take = [&](size_t bytes) -> char* {
    char* p = ws + off;
    off = (off + bytes + 255) & ~(size_t)255;
    return p;
  };
  bf16_t* xb    = (bf16_t*)take((size_t)T_TOK * DIM * 2);
  bf16_t* w1t   = (bf16_t*)take((size_t)NEXP * HEXP * DIM * 2);
  bf16_t* w3t   = (bf16_t*)take((size_t)NEXP * HEXP * DIM * 2);
  bf16_t* w2t   = (bf16_t*)take((size_t)NEXP * DIM * HEXP * 2);
  bf16_t* ws1t  = (bf16_t*)take((size_t)HSH * DIM * 2);
  bf16_t* ws3t  = (bf16_t*)take((size_t)HSH * DIM * 2);
  bf16_t* ws2t  = (bf16_t*)take((size_t)DIM * HSH * 2);
  bf16_t* hs    = (bf16_t*)take((size_t)T_TOK * HSH * 2);          // shared H
  bf16_t* hr    = (bf16_t*)take((size_t)T_TOK * 2 * HEXP * 2);     // 4096 rows
  float*  ybuf  = (float*) take((size_t)T_TOK * 2 * DIM * 4);      // 4096 rows
  int*    tidx  = (int*)   take((size_t)T_TOK * 2 * 4);
  float*  tw    = (float*) take((size_t)T_TOK * 2 * 4);
  int*    tokl  = (int*)   take((size_t)T_TOK * 2 * 4);
  float*  wgtl  = (float*) take((size_t)T_TOK * 2 * 4);
  int*    rowof = (int*)   take((size_t)T_TOK * 2 * 4);
  int*    counts= (int*)   take(256);
  int*    offs  = (int*)   take(256);

  dim3 tb(32, 8, 1);

  // 1) activations fp32 -> bf16
  convert_f32_to_bf16<<<(T_TOK * DIM) / (256 * 4), 256, 0, stream>>>(
      x, xb, T_TOK * DIM);

  // 2) weights fp32 -> bf16 transposed to [N][K]
  transpose_f32_to_bf16<<<dim3(HEXP / 32, DIM / 32, NEXP), tb, 0, stream>>>(
      w1, w1t, DIM, HEXP);
  transpose_f32_to_bf16<<<dim3(HEXP / 32, DIM / 32, NEXP), tb, 0, stream>>>(
      w3, w3t, DIM, HEXP);
  transpose_f32_to_bf16<<<dim3(DIM / 32, HEXP / 32, NEXP), tb, 0, stream>>>(
      w2, w2t, HEXP, DIM);
  transpose_f32_to_bf16<<<dim3(HSH / 32, DIM / 32, 1), tb, 0, stream>>>(
      ws1, ws1t, DIM, HSH);
  transpose_f32_to_bf16<<<dim3(HSH / 32, DIM / 32, 1), tb, 0, stream>>>(
      ws3, ws3t, DIM, HSH);
  transpose_f32_to_bf16<<<dim3(DIM / 32, HSH / 32, 1), tb, 0, stream>>>(
      ws2, ws2t, HSH, DIM);

  // 3) router + deterministic compaction
  router_kernel<<<T_TOK / 8, 256, 0, stream>>>(x, gate_w, tidx, tw);
  assign_kernel<<<1, 32, 0, stream>>>(tidx, tw, tokl, wgtl, rowof, counts, offs);

  // 4) shared expert: up (silu(X@ws1)*(X@ws3)) then down -> writes out
  moe_up_kernel<<<dim3(T_TOK / 64, HSH / 256, 1), 256, 0, stream>>>(
      xb, ws1t, ws3t, hs, nullptr, nullptr, nullptr, DIM, HSH);
  moe_down_kernel<<<dim3(T_TOK / 64, DIM / 256, 1), 256, 0, stream>>>(
      hs, ws2t, out, nullptr, nullptr, nullptr, HSH, DIM);

  // 5) routed experts over packed token lists (blocks early-exit past count)
  moe_up_kernel<<<dim3(T_TOK / 64, HEXP / 256, NEXP), 256, 0, stream>>>(
      xb, w1t, w3t, hr, tokl, counts, offs, DIM, HEXP);
  moe_down_kernel<<<dim3(T_TOK / 64, DIM / 256, NEXP), 256, 0, stream>>>(
      hr, w2t, ybuf, counts, offs, wgtl, HEXP, DIM);

  // 6) out[t] += weighted routed contributions (fixed order, deterministic)
  combine_kernel<<<(T_TOK * DIM / 4) / 256, 256, 0, stream>>>(ybuf, rowof, out);
}
